// SparseWeightedConv2D_54812372631846
// MI455X (gfx1250) — matrix operand order repaired and checked by reading, not compiled, not run on hardware
//
#include <hip/hip_runtime.h>

// ---------------------------------------------------------------------------
// Sparse-weight conv2d as im2col GEMM on gfx1250 (MI455X).
//   M = OUT_CH = 256, N = B*H*W = 25088, K = IN_CH*3*3 = 2304.
// bf16 WMMA (v_wmma_f32_16x16x32_bf16) with fp32 accumulation.
// Prepass converts fp32 w -> bf16 and x -> zero-padded bf16 [8][256][58][58]
// in workspace, so the GEMM's im2col fill has NO bounds checks (pure
// add + global_load_u16). k -> (c,kh,kw) offsets are precomputed in an LDS
// table (one ds_load_b128 per wave per K-step instead of division chains).
// Weight tile streamed with global_load_async_to_lds_b128 (ASYNCcnt).
// ---------------------------------------------------------------------------

typedef __attribute__((ext_vector_type(16))) __bf16          v16bf;
typedef __attribute__((ext_vector_type(8)))  float           v8f;
typedef __attribute__((ext_vector_type(8)))  unsigned short  v8u16;
typedef __attribute__((ext_vector_type(16))) unsigned short  v16u16;
typedef int v4i __attribute__((vector_size(16)));
typedef __attribute__((address_space(1))) v4i* gas_v4i_p;   // global AS ptr
typedef __attribute__((address_space(3))) v4i* las_v4i_p;   // LDS AS ptr

#define IN_CH   256
#define OUT_CH  256
#define HW      56
#define SP      (HW * HW)        // 3136
#define HWP     58               // padded spatial extent
#define SPP     (HWP * HWP)      // 3364
#define BATCH   8
#define KTOT    (IN_CH * 9)      // 2304
#define NTOT    (BATCH * SP)     // 25088

#define TILE_M  128
#define TILE_N  128
#define TILE_K  32
#define KSTEPS  (KTOT / TILE_K)  // 72
#define LDA     40               // padded row stride (bf16 elems): 80B
#define LDB     40

#if defined(__has_builtin)
#if __has_builtin(__builtin_amdgcn_global_load_async_to_lds_b128)
#define HAVE_ASYNC_LDS 1
#endif
#endif

__device__ __forceinline__ unsigned short f2bf(float f) {
  unsigned u = __float_as_uint(f);
  u += 0x7fffu + ((u >> 16) & 1u);     // round-to-nearest-even
  return (unsigned short)(u >> 16);
}

// fp32 -> bf16, 4 elements/thread (weights)
__global__ void cvt_bf16_kernel(const float* __restrict__ in,
                                unsigned short* __restrict__ out, int n4) {
  int i = blockIdx.x * blockDim.x + threadIdx.x;
  if (i >= n4) return;
  const float4 f = ((const float4*)in)[i];
  ushort4 o;
  o.x = f2bf(f.x); o.y = f2bf(f.y); o.z = f2bf(f.z); o.w = f2bf(f.w);
  ((ushort4*)out)[i] = o;
}

// fp32 x[8][256][56][56] -> bf16 xpad[8][256][58][58] with zero halo
__global__ void cvt_pad_x_kernel(const float* __restrict__ in,
                                 unsigned short* __restrict__ out, int n) {
  int i = blockIdx.x * blockDim.x + threadIdx.x;
  if (i >= n) return;
  unsigned bc = (unsigned)i / SPP;
  unsigned p  = (unsigned)i - bc * SPP;
  unsigned hp = p / HWP;
  unsigned wp = p - hp * HWP;
  unsigned short v = 0;
  if (hp >= 1u && hp <= (unsigned)HW && wp >= 1u && wp <= (unsigned)HW)
    v = f2bf(in[bc * SP + (hp - 1u) * HW + (wp - 1u)]);
  out[i] = v;
}

__global__ __launch_bounds__(256)
void sparse_conv_gemm(const unsigned short* __restrict__ wbf,   // [256][2304] bf16
                      const unsigned short* __restrict__ xpad,  // [8][256][58][58] bf16
                      float* __restrict__ out) {                // [8][256][56][56] f32
  __shared__ unsigned short As[2][TILE_M * LDA];
  __shared__ unsigned short Bs[2][TILE_N * LDB];   // N-major, K contiguous
  __shared__ int Koff[KTOT];                       // k -> c*SPP + kh*HWP + kw

  const unsigned tid  = threadIdx.x;
  const unsigned wave = tid >> 5;
  const unsigned lane = tid & 31u;
  const unsigned hf   = lane >> 4;
  const unsigned l15  = lane & 15u;
  const unsigned wm   = wave >> 2;     // 0..1 -> 64 output rows each
  const unsigned wn   = wave & 3u;     // 0..3 -> 32 output cols each

  const unsigned m0 = blockIdx.y * TILE_M;
  const unsigned n0 = blockIdx.x * TILE_N;

  // ---- fill the k-offset table: thread t owns k = t*9 .. t*9+8 (c == t)
  {
    const int kb = (int)tid * 9;
    const int cb = (int)tid * SPP;
#pragma unroll
    for (int ii = 0; ii < 9; ++ii)
      Koff[kb + ii] = cb + (ii / 3) * HWP + (ii % 3);
  }

  // ---- per-thread column (n) base into padded x (fixed across K loop)
  int xbase[4];
#pragma unroll
  for (int j = 0; j < 4; ++j) {
    unsigned n  = n0 + lane + 32u * j;
    unsigned b  = n / SP;
    unsigned p  = n - b * SP;
    unsigned oh = p / HW;
    unsigned ow = p - oh * HW;
    // padded coords: (oh + kh, ow + kw); kh/kw live inside Koff[]
    xbase[j] = (int)(b * (IN_CH * SPP)) + (int)(oh * HWP + ow);
  }

  // ---- A-tile fill indices: 2 threads per row, 16 bf16 (32B) each
  const unsigned am = tid >> 1;            // 0..127
  const unsigned ak = (tid & 1u) * 16u;    // 0 or 16

#ifndef HAVE_ASYNC_LDS
  uint4 ra0, ra1;                          // reg-staged weight chunk
#endif
  unsigned rb[4][2];                       // per n: 4 bf16 over 4 k-columns

  auto prefetchA = [&](int buf, int ks) {
    const unsigned k0 = (unsigned)ks * TILE_K;
    const unsigned short* gp = wbf + (size_t)(m0 + am) * KTOT + k0 + ak;
    unsigned short* lp = &As[buf][am * LDA + ak];
#ifdef HAVE_ASYNC_LDS
    __builtin_amdgcn_global_load_async_to_lds_b128(
        (gas_v4i_p)gp, (las_v4i_p)lp, 0, 0);
    __builtin_amdgcn_global_load_async_to_lds_b128(
        (gas_v4i_p)(gp + 8), (las_v4i_p)(lp + 8), 0, 0);
#else
    const uint4* wp = (const uint4*)gp;
    ra0 = wp[0];
    ra1 = wp[1];
    (void)lp;
#endif
  };

  auto loadB = [&](int ks) {
    const unsigned k0 = (unsigned)ks * TILE_K;
    const int4 kof = *(const int4*)&Koff[k0 + wave * 4u];
    const int ka4[4] = {kof.x, kof.y, kof.z, kof.w};
#pragma unroll
    for (int ki = 0; ki < 4; ++ki) {
#pragma unroll
      for (int j = 0; j < 4; ++j) {
        unsigned short v = xpad[xbase[j] + ka4[ki]];   // no bounds check
        if      (ki == 0) rb[j][0]  = v;
        else if (ki == 1) rb[j][0] |= (unsigned)v << 16;
        else if (ki == 2) rb[j][1]  = v;
        else              rb[j][1] |= (unsigned)v << 16;
      }
    }
  };

  auto storeTile = [&](int buf) {
#ifndef HAVE_ASYNC_LDS
    uint4* as = (uint4*)&As[buf][am * LDA + ak];
    as[0] = ra0;
    as[1] = ra1;
#endif
#pragma unroll
    for (int j = 0; j < 4; ++j) {
      uint2 d;
      d.x = rb[j][0];
      d.y = rb[j][1];
      *(uint2*)&Bs[buf][(lane + 32u * j) * LDB + wave * 4u] = d;
    }
  };

  auto waitAsync = [&]() {
#ifdef HAVE_ASYNC_LDS
#if __has_builtin(__builtin_amdgcn_s_wait_asynccnt)
    __builtin_amdgcn_s_wait_asynccnt(0);
#else
    asm volatile("s_wait_asynccnt 0" ::: "memory");
#endif
#endif
  };

  v8f acc[4][2];
#pragma unroll
  for (int i = 0; i < 4; ++i)
#pragma unroll
    for (int j = 0; j < 2; ++j)
      acc[i][j] = (v8f){0.f, 0.f, 0.f, 0.f, 0.f, 0.f, 0.f, 0.f};

  auto compute = [&](int buf) {
    v16bf af[4], bfr[2];
#pragma unroll
    for (int i = 0; i < 4; ++i) {
      // A lane layout: row M = l15, K = {hf*8..+7, 16+hf*8..+7}
      const unsigned short* ap = &As[buf][(wm * 64u + i * 16u + l15) * LDA];
      v8u16 lo = *(const v8u16*)(ap + hf * 8u);
      v8u16 hi = *(const v8u16*)(ap + 16u + hf * 8u);
      v16u16 cb = __builtin_shufflevector(lo, hi, 0, 1, 2, 3, 4, 5, 6, 7,
                                          8, 9, 10, 11, 12, 13, 14, 15);
      af[i] = __builtin_bit_cast(v16bf, cb);
    }
#pragma unroll
    for (int j = 0; j < 2; ++j) {
      // B lane layout: col N = l15, K = {hf*8..+7, 16+hf*8..+7} (N-major LDS)
      const unsigned short* bp =
          &Bs[buf][(wn * 32u + j * 16u + l15) * LDB];
      v8u16 lo = *(const v8u16*)(bp + hf * 8u);
      v8u16 hi = *(const v8u16*)(bp + 16u + hf * 8u);
      v16u16 cb = __builtin_shufflevector(lo, hi, 0, 1, 2, 3, 4, 5, 6, 7,
                                          8, 9, 10, 11, 12, 13, 14, 15);
      bfr[j] = __builtin_bit_cast(v16bf, cb);
    }
#pragma unroll
    for (int i = 0; i < 4; ++i)
#pragma unroll
      for (int j = 0; j < 2; ++j)
        acc[i][j] = __builtin_amdgcn_wmma_f32_16x16x32_bf16(
            false, af[i], false, bfr[j], (short)0, acc[i][j], false, false);
  };

  // ---- software-pipelined, double-buffered K loop
  __syncthreads();                 // Koff table ready
  prefetchA(0, 0);
  loadB(0);
  storeTile(0);
  waitAsync();
#pragma unroll 2
  for (int ks = 0; ks < KSTEPS; ++ks) {
    const int cur = ks & 1;
    __syncthreads();               // buf 'cur' published; buf cur^1 fully consumed
    if (ks + 1 < KSTEPS) {
      prefetchA(cur ^ 1, ks + 1);  // async: writes LDS directly; reg: global->reg
      loadB(ks + 1);
    }
    compute(cur);
    if (ks + 1 < KSTEPS) storeTile(cur ^ 1);
    waitAsync();                   // A-tile async writes done before next barrier
  }

  // ---- epilogue: C/D layout -> out[b][o][h][w] f32
#pragma unroll
  for (int i = 0; i < 4; ++i) {
    const unsigned Mg = m0 + wm * 64u + i * 16u + hf * 8u;
#pragma unroll
    for (int j = 0; j < 2; ++j) {
      const unsigned N = n0 + wn * 32u + j * 16u + l15;
      const unsigned b = N / SP;
      const unsigned p = N - b * SP;
      float* op = out + ((size_t)(b * OUT_CH + Mg) * SP + p);
#pragma unroll
      for (int g = 0; g < 8; ++g)
        op[(size_t)g * SP] = acc[i][j][g];
    }
  }
}

extern "C" void kernel_launch(void* const* d_in, const int* in_sizes, int n_in,
                              void* d_out, int out_size, void* d_ws, size_t ws_size,
                              hipStream_t stream) {
  const float* x = (const float*)d_in[0];   // [8][256][56][56]
  const float* w = (const float*)d_in[1];   // [256][2304]

  unsigned short* wbf  = (unsigned short*)d_ws;
  unsigned short* xpad = wbf + (size_t)OUT_CH * KTOT;   // +1,179,648 B

  const int wN4 = (OUT_CH * KTOT) / 4;            // 147456
  const int xNp = BATCH * IN_CH * SPP;            // 6,889,472 padded elems
  cvt_bf16_kernel<<<(wN4 + 255) / 256, 256, 0, stream>>>(w, wbf, wN4);
  cvt_pad_x_kernel<<<(xNp + 255) / 256, 256, 0, stream>>>(x, xpad, xNp);

  dim3 grid(NTOT / TILE_N, OUT_CH / TILE_M, 1);   // (196, 2)
  sparse_conv_gemm<<<grid, 256, 0, stream>>>(wbf, xpad, (float*)d_out);
}
